// MultiHeadAttention_10943576670946
// MI455X (gfx1250) — matrix operand order repaired and checked by reading, not compiled, not run on hardware
//
#include <hip/hip_runtime.h>
#include <hip/hip_bf16.h>

// ---------------------------------------------------------------------------
// MHA forward for MI455X (gfx1250, wave32, WMMA + TDM).
//   B=2, S=4096, D_MODEL=512, H=8, D_HEAD=64
// Outputs (concat in d_out): context [2,4096,512] fp32, attn [16,4096,4096] fp32
// Workspace: Q/K/V f16 projections, 8192*512*2B each = 24 MB total.
// ---------------------------------------------------------------------------

typedef __attribute__((ext_vector_type(16))) _Float16 v16h;
typedef __attribute__((ext_vector_type(8)))  float    v8f;

typedef unsigned int u32;
typedef __attribute__((ext_vector_type(4))) u32 u32x4;
typedef __attribute__((ext_vector_type(8))) int i32x8;
typedef __attribute__((ext_vector_type(4))) int i32x4;

#if __has_builtin(__builtin_amdgcn_tensor_load_to_lds) && \
    __has_builtin(__builtin_amdgcn_s_wait_tensorcnt)
#define HAVE_TDM 1
#else
#define HAVE_TDM 0
#endif

union FragU { v16h v; uint4 q[2]; };

// A-matrix fragment (16x32 f16), ISA 7.12.2 layout:
// lane<16: row M=lane, halves 0..7 = K 0..7,  halves 8..15 = K 16..23
// lane>=16: row M=lane-16, halves = K 8..15 and K 24..31
__device__ __forceinline__ v16h frag_a(const _Float16* p, int ldk) {
    const int lane = threadIdx.x & 31;
    const int r  = lane & 15;
    const int kb = (lane >> 4) << 3;     // 0 or 8
    FragU f;
    f.q[0] = *(const uint4*)(p + r * ldk + kb);
    f.q[1] = *(const uint4*)(p + r * ldk + kb + 16);
    return f.v;
}

// B-matrix fragment (32x16 f16), stored in LDS transposed as [N][K]:
// lane<16: col N=lane, halves = K 0..7 and K 16..23 ; lane>=16: K 8..15 and K 24..31
__device__ __forceinline__ v16h frag_b(const _Float16* p, int ldk) {
    const int lane = threadIdx.x & 31;
    const int n  = lane & 15;
    const int kb = (lane >> 4) << 3;     // 0 or 8
    FragU f;
    f.q[0] = *(const uint4*)(p + n * ldk + kb);
    f.q[1] = *(const uint4*)(p + n * ldk + kb + 16);
    return f.v;
}

#define WMMA_F16(A, Bm, C) \
    __builtin_amdgcn_wmma_f32_16x16x32_f16(false, (A), false, (Bm), (short)0, (C), false, false)

#if HAVE_TDM
// Issue one TDM 2D-tile load: f16 tile [tile_h rows x tile_w elems], global row
// stride row_stride elems, into contiguous LDS at lds_dst. D# per ISA 8.3/8.4.
// 6-arg builtin form (clang-23 / therock-10.0): trailing i32x8 + cpol.
__device__ __forceinline__ void tdm_load_tile_f16(
    const _Float16* gsrc, void* lds_dst, int tile_w, int tile_h, int row_stride)
{
    const unsigned long long ga = (unsigned long long)(size_t)gsrc;
    const u32 lds = (u32)(size_t)lds_dst;        // low 32 bits = LDS byte addr
    u32x4 g0;
    g0.x = 1u;                                   // count=1 (valid), load, user
    g0.y = lds;                                  // lds_addr
    g0.z = (u32)(ga & 0xFFFFFFFFu);              // global_addr[31:0]
    g0.w = (u32)((ga >> 32) & 0x1FFFFFFu)        // global_addr[56:32]
         | (2u << 30);                           // type=2 ("image")
    i32x8 g1;
    g1[0] = (1 << 16);                           // wg_mask=0, data_size=1 (2B)
    g1[1] = (tile_w & 0xFFFF) << 16;             // tensor_dim0[15:0]
    g1[2] = ((tile_w >> 16) & 0xFFFF)            // tensor_dim0[31:16]
          | ((tile_h & 0xFFFF) << 16);           // tensor_dim1[15:0]
    g1[3] = ((tile_h >> 16) & 0xFFFF)            // tensor_dim1[31:16]
          | ((tile_w & 0xFFFF) << 16);           // tile_dim0
    g1[4] = (tile_h & 0xFFFF);                   // tile_dim1 (tile_dim2=0)
    g1[5] = row_stride;                          // tensor_dim0_stride[31:0]
    g1[6] = 0;                                   // stride0[47:32], stride1[15:0]
    g1[7] = 0;
    const i32x4 gz4 = {0, 0, 0, 0};              // 2D tensor: groups 2/3 unused
    const i32x8 gz8 = {0, 0, 0, 0, 0, 0, 0, 0};
    __builtin_amdgcn_tensor_load_to_lds(g0, g1, gz4, gz4, gz8, 0);
}
#endif

// ---------------------------------------------------------------------------
// Kernel 1: QKV projection GEMM.  Y = X @ W + b  (fp32 in -> f16 out)
// M=8192, N=512, K=512.  grid (128, 8, 3), block 128 (4 waves).
// Block tile 64x64, per-wave 32x32 (2x2 WMMA tiles).
// ---------------------------------------------------------------------------
__global__ __launch_bounds__(128) void mha_proj(
    const float* __restrict__ xq, const float* __restrict__ xk, const float* __restrict__ xv,
    const float* __restrict__ wqm, const float* __restrict__ wkm, const float* __restrict__ wvm,
    const float* __restrict__ bqv, const float* __restrict__ bkv, const float* __restrict__ bvv,
    _Float16* __restrict__ oq, _Float16* __restrict__ ok, _Float16* __restrict__ ov)
{
    const float* X; const float* W; const float* bias; _Float16* O;
    switch (blockIdx.z) {
        case 0:  X = xq; W = wqm; bias = bqv; O = oq; break;
        case 1:  X = xk; W = wkm; bias = bkv; O = ok; break;
        default: X = xv; W = wvm; bias = bvv; O = ov; break;
    }
    const int m0 = blockIdx.x * 64;
    const int n0 = blockIdx.y * 64;
    const int t  = threadIdx.x;
    const int w  = t >> 5, lane = t & 31;
    const int wr = w >> 1, wc = w & 1;

    __shared__ __align__(16) _Float16 As[64][32];      // [m][k]
    __shared__ __align__(16) _Float16 Bs[64][32];      // [n][k]  (W transposed)

    v8f acc[2][2];
#pragma unroll
    for (int i = 0; i < 2; ++i)
#pragma unroll
        for (int j = 0; j < 2; ++j) acc[i][j] = (v8f){0.f,0.f,0.f,0.f,0.f,0.f,0.f,0.f};

    for (int k0 = 0; k0 < 512; k0 += 32) {
        // stage A: 64x32 fp32 -> f16   (512 float4, 4 per thread)
#pragma unroll
        for (int i = 0; i < 4; ++i) {
            const int jj = t + i * 128;
            const int r = jj >> 3, c = (jj & 7) << 2;
            float4 v = *(const float4*)(X + (size_t)(m0 + r) * 512 + k0 + c);
            As[r][c + 0] = (_Float16)v.x;  As[r][c + 1] = (_Float16)v.y;
            As[r][c + 2] = (_Float16)v.z;  As[r][c + 3] = (_Float16)v.w;
        }
        // stage B^T: W rows k0..k0+31, cols n0..n0+63 -> Bs[n][k]
#pragma unroll
        for (int i = 0; i < 4; ++i) {
            const int jj = t + i * 128;
            const int kk = jj >> 4, c = (jj & 15) << 2;
            float4 v = *(const float4*)(W + (size_t)(k0 + kk) * 512 + n0 + c);
            Bs[c + 0][kk] = (_Float16)v.x;  Bs[c + 1][kk] = (_Float16)v.y;
            Bs[c + 2][kk] = (_Float16)v.z;  Bs[c + 3][kk] = (_Float16)v.w;
        }
        __syncthreads();
        v16h a0 = frag_a(&As[wr * 32 +  0][0], 32);
        v16h a1 = frag_a(&As[wr * 32 + 16][0], 32);
        v16h b0 = frag_b(&Bs[wc * 32 +  0][0], 32);
        v16h b1 = frag_b(&Bs[wc * 32 + 16][0], 32);
        acc[0][0] = WMMA_F16(a0, b0, acc[0][0]);
        acc[0][1] = WMMA_F16(a0, b1, acc[0][1]);
        acc[1][0] = WMMA_F16(a1, b0, acc[1][0]);
        acc[1][1] = WMMA_F16(a1, b1, acc[1][1]);
        __syncthreads();
    }

    const int col = lane & 15;
    const int rb  = (lane >> 4) << 3;
#pragma unroll
    for (int i = 0; i < 2; ++i)
#pragma unroll
        for (int j = 0; j < 2; ++j) {
            const int n = n0 + wc * 32 + j * 16 + col;
            const float bv = bias[n];
#pragma unroll
            for (int r = 0; r < 8; ++r) {
                const int row = m0 + wr * 32 + i * 16 + rb + r;
                O[(size_t)row * 512 + n] = (_Float16)(acc[i][j][r] + bv);
            }
        }
}

// ---------------------------------------------------------------------------
// Kernel 2: scores = (Q @ K^T) * 0.125 per (h,b).  M=N=4096, K=64.
// grid (32, 32, 16), block 256 (8 waves). Block tile 128x128; wave 32x64.
// Q/K tiles staged by the Tensor Data Mover (one 2D-tile DMA each).
// ---------------------------------------------------------------------------
__global__ __launch_bounds__(256) void mha_scores(
    const _Float16* __restrict__ Qh, const _Float16* __restrict__ Kh,
    float* __restrict__ attn)
{
    const int z = blockIdx.z;           // z = h*2 + b  (== attn index h*B+b)
    const int h = z >> 1, b = z & 1;
    const int m0 = blockIdx.x * 128;
    const int n0 = blockIdx.y * 128;
    const int t = threadIdx.x, w = t >> 5, lane = t & 31;
    const int wr = w & 3;               // m offset wr*32
    const int wc = w >> 2;              // n offset wc*64

    __shared__ __align__(16) _Float16 Qt[128][64];
    __shared__ __align__(16) _Float16 Kt[128][64];

    const size_t qbase = ((size_t)(b * 4096 + m0)) * 512 + h * 64;
    const size_t kbase = ((size_t)(b * 4096 + n0)) * 512 + h * 64;

#if HAVE_TDM
    if (w == 0) {                       // one wave issues both tile DMAs
        tdm_load_tile_f16(Qh + qbase, &Qt[0][0], 64, 128, 512);
        tdm_load_tile_f16(Kh + kbase, &Kt[0][0], 64, 128, 512);
        __builtin_amdgcn_s_wait_tensorcnt(0);
    }
    asm volatile("" ::: "memory");      // LDS written behind compiler's back
    __syncthreads();
#else
#pragma unroll
    for (int i = 0; i < 4; ++i) {
        const int jj = t + i * 256;
        const int r = jj >> 3, c = (jj & 7) << 3;
        *(uint4*)&Qt[r][c] = *(const uint4*)(Qh + qbase + (size_t)r * 512 + c);
        *(uint4*)&Kt[r][c] = *(const uint4*)(Kh + kbase + (size_t)r * 512 + c);
    }
    __syncthreads();
#endif

    v8f acc[2][4];
#pragma unroll
    for (int i = 0; i < 2; ++i)
#pragma unroll
        for (int j = 0; j < 4; ++j) acc[i][j] = (v8f){0.f,0.f,0.f,0.f,0.f,0.f,0.f,0.f};

#pragma unroll
    for (int kk = 0; kk < 64; kk += 32) {
        v16h a0 = frag_a(&Qt[wr * 32 +  0][kk], 64);
        v16h a1 = frag_a(&Qt[wr * 32 + 16][kk], 64);
#pragma unroll
        for (int j = 0; j < 4; ++j) {
            v16h bm = frag_b(&Kt[wc * 64 + j * 16][kk], 64);
            acc[0][j] = WMMA_F16(a0, bm, acc[0][j]);
            acc[1][j] = WMMA_F16(a1, bm, acc[1][j]);
        }
    }

    const int col = lane & 15;
    const int rb  = (lane >> 4) << 3;
#pragma unroll
    for (int i = 0; i < 2; ++i)
#pragma unroll
        for (int j = 0; j < 4; ++j)
#pragma unroll
            for (int r = 0; r < 8; ++r) {
                const int row = m0 + wr * 32 + i * 16 + rb + r;
                const int cn  = n0 + wc * 64 + j * 16 + col;
                // 1 GB stream, reuse distance >> L2: non-temporal store
                __builtin_nontemporal_store(acc[i][j][r] * 0.125f,
                    attn + ((size_t)z * 4096 + row) * 4096 + cn);
            }
}

// ---------------------------------------------------------------------------
// Kernel 3: in-place row softmax over attn.  grid 65536, block 256.
// ---------------------------------------------------------------------------
__global__ __launch_bounds__(256) void mha_softmax(float* __restrict__ attn)
{
    float* p = attn + (size_t)blockIdx.x * 4096;
    const int t = threadIdx.x;
    __shared__ float red[256];

    float v[16];
    float lmax = -3.402823466e38f;
#pragma unroll
    for (int i = 0; i < 16; ++i) {
        v[i] = __builtin_nontemporal_load(p + t + i * 256);
        lmax = fmaxf(lmax, v[i]);
    }
    red[t] = lmax; __syncthreads();
    for (int s = 128; s > 0; s >>= 1) {
        if (t < s) red[t] = fmaxf(red[t], red[t + s]);
        __syncthreads();
    }
    const float m = red[0]; __syncthreads();

    float lsum = 0.f;
#pragma unroll
    for (int i = 0; i < 16; ++i) {
        v[i] = __expf(v[i] - m);
        lsum += v[i];
    }
    red[t] = lsum; __syncthreads();
    for (int s = 128; s > 0; s >>= 1) {
        if (t < s) red[t] += red[t + s];
        __syncthreads();
    }
    const float inv = 1.0f / red[0];
#pragma unroll
    for (int i = 0; i < 16; ++i)
        __builtin_nontemporal_store(v[i] * inv, p + t + i * 256);
}

// ---------------------------------------------------------------------------
// Kernel 4: context = P @ V per (h,b).  M=4096, N=64, K=4096.
// grid (32, 1, 16), block 256 (8 waves). Block tile 128x64; wave 16x64.
// P fp32 -> f16 in LDS on the fly (NT loads: P is a 1 GB single-use stream).
// ---------------------------------------------------------------------------
__global__ __launch_bounds__(256) void mha_context(
    const float* __restrict__ attn, const _Float16* __restrict__ Vh,
    float* __restrict__ out)
{
    const int z = blockIdx.z;
    const int h = z >> 1, b = z & 1;
    const int m0 = blockIdx.x * 128;
    const int t = threadIdx.x, w = t >> 5, lane = t & 31;

    __shared__ __align__(16) _Float16 At[128][32];   // [m][k]
    __shared__ __align__(16) _Float16 Vt[64][32];    // [n][k]  (V transposed)

    v8f acc[4];
#pragma unroll
    for (int j = 0; j < 4; ++j) acc[j] = (v8f){0.f,0.f,0.f,0.f,0.f,0.f,0.f,0.f};

    const float* arow = attn + ((size_t)z * 4096 + m0) * 4096;
    const _Float16* vbase = Vh + (size_t)(b * 4096) * 512 + h * 64;

    for (int k0 = 0; k0 < 4096; k0 += 32) {
        // stage P: 128x32 fp32 -> f16  (1024 float4, 4 per thread)
#pragma unroll
        for (int i = 0; i < 4; ++i) {
            const int jj = t + i * 256;
            const int r = jj >> 3, c = (jj & 7) << 2;
            const float* src = arow + (size_t)r * 4096 + k0 + c;
            float4 v;
            v.x = __builtin_nontemporal_load(src + 0);
            v.y = __builtin_nontemporal_load(src + 1);
            v.z = __builtin_nontemporal_load(src + 2);
            v.w = __builtin_nontemporal_load(src + 3);
            At[r][c + 0] = (_Float16)v.x;  At[r][c + 1] = (_Float16)v.y;
            At[r][c + 2] = (_Float16)v.z;  At[r][c + 3] = (_Float16)v.w;
        }
        // stage V^T: rows k0..k0+31, 64 head cols -> Vt[n][k] (1 uint4/thread)
        {
            const int kk = t >> 3;
            const int c  = (t & 7) << 3;
            _Float16 tmp[8];
            *(uint4*)tmp = *(const uint4*)(vbase + (size_t)(k0 + kk) * 512 + c);
#pragma unroll
            for (int e = 0; e < 8; ++e) Vt[c + e][kk] = tmp[e];
        }
        __syncthreads();
        v16h a = frag_a(&At[w * 16][0], 32);
#pragma unroll
        for (int j = 0; j < 4; ++j) {
            v16h bm = frag_b(&Vt[j * 16][0], 32);
            acc[j] = WMMA_F16(a, bm, acc[j]);
        }
        __syncthreads();
    }

    const int col = lane & 15;
    const int rb  = (lane >> 4) << 3;
#pragma unroll
    for (int j = 0; j < 4; ++j)
#pragma unroll
        for (int r = 0; r < 8; ++r) {
            const int row = m0 + w * 16 + rb + r;
            out[((size_t)(b * 4096 + row)) * 512 + h * 64 + j * 16 + col] = acc[j][r];
        }
}

// ---------------------------------------------------------------------------
extern "C" void kernel_launch(void* const* d_in, const int* in_sizes, int n_in,
                              void* d_out, int out_size, void* d_ws, size_t ws_size,
                              hipStream_t stream) {
    (void)in_sizes; (void)n_in; (void)out_size; (void)ws_size;
    const float* query = (const float*)d_in[0];
    const float* key   = (const float*)d_in[1];
    const float* value = (const float*)d_in[2];
    const float* Wq    = (const float*)d_in[3];
    const float* bq    = (const float*)d_in[4];
    const float* Wk    = (const float*)d_in[5];
    const float* bk    = (const float*)d_in[6];
    const float* Wv    = (const float*)d_in[7];
    const float* bv    = (const float*)d_in[8];

    float* ctx_out  = (float*)d_out;                          // [2,4096,512]
    float* attn_out = ctx_out + (size_t)2 * 4096 * 512;       // [16,4096,4096]

    _Float16* Qh = (_Float16*)d_ws;                           // [8192,512] each
    _Float16* Kh = Qh + (size_t)8192 * 512;
    _Float16* Vh = Kh + (size_t)8192 * 512;

    // 1) QKV projections (fp32 -> f16)
    mha_proj<<<dim3(8192 / 64, 512 / 64, 3), 128, 0, stream>>>(
        query, key, value, Wq, Wk, Wv, bq, bk, bv, Qh, Kh, Vh);

    // 2) raw scaled scores -> attn region (TDM-staged tiles)
    mha_scores<<<dim3(4096 / 128, 4096 / 128, 16), 256, 0, stream>>>(Qh, Kh, attn_out);

    // 3) row softmax in place
    mha_softmax<<<dim3(16 * 4096), 256, 0, stream>>>(attn_out);

    // 4) context = P @ V
    mha_context<<<dim3(4096 / 128, 1, 16), 256, 0, stream>>>(attn_out, Vh, ctx_out);
}
